// EnsembleFormer_62646392979780
// MI455X (gfx1250) — matrix operand order repaired and checked by reading, not compiled, not run on hardware
//
#include <hip/hip_runtime.h>
#include <hip/hip_bf16.h>

// ---------------------------------------------------------------------------
// EnsembleFormer collapsed pipeline for MI455X (gfx1250, wave32).
//   K1: pool x 64x64 -> 8x8 agents            (reads 100.7 MB, float4)
//   K2: WMMA GEMM  x8(1024x384) @ W1(384x480) + b1 -> y8
//   K3: clustering (hard + soft) + cosine-softmax mixing -> out8(1024x96)
//   K4: WMMA GEMM  out8(1024x96) @ W2(96x384) -> z8
//   K5: bilinear upsample 8x8 -> 64x64 + b2   (writes 100.7 MB, b128 stores)
// LDS tiles for the GEMMs are stored in WMMA *fragment order* so each lane
// fetches its v16h operand with two ds_load_b128 instead of 16 ds_load_u16.
// ---------------------------------------------------------------------------

typedef __attribute__((ext_vector_type(16))) _Float16 v16h;
typedef __attribute__((ext_vector_type(2)))  _Float16 h2;
typedef __attribute__((ext_vector_type(8)))  float    v8f;

#define HDIM 24          // per-head channels
#define NSD  480         // NSTREAM * D
#define DDIM 96          // HEADS * HD
#define CIN  384

// ---------------- K1: 64x64 -> 8x8 average pool, one block per (b,cin) plane
__global__ __launch_bounds__(256)
void pool_x_kernel(const float* __restrict__ x, float* __restrict__ x8) {
    __shared__ float part[64][8];
    const int plane = blockIdx.x;                 // b*CIN + c
    const float* __restrict__ src = x + (size_t)plane * 4096;
    const int tid = threadIdx.x;
    const int r = tid >> 2;                       // row 0..63
    const int q = tid & 3;                        // 16-col quarter
    const float4* p = (const float4*)(src + r * 64 + q * 16);
    float4 v0 = p[0], v1 = p[1], v2 = p[2], v3 = p[3];
    part[r][2 * q]     = v0.x + v0.y + v0.z + v0.w + v1.x + v1.y + v1.z + v1.w;
    part[r][2 * q + 1] = v2.x + v2.y + v2.z + v2.w + v3.x + v3.y + v3.z + v3.w;
    __syncthreads();
    if (tid < 64) {
        const int oy = tid >> 3, ox = tid & 7;
        float s = 0.f;
#pragma unroll
        for (int i = 0; i < 8; ++i) s += part[oy * 8 + i][ox];
        const int b = plane / CIN;
        const int c = plane - b * CIN;
        x8[((size_t)b * 64 + tid) * CIN + c] = s * (1.0f / 64.0f);
    }
}

// ---------------- K2/K4: WMMA f16 GEMM, C = A(MxK) @ B(KxN) [+ bias(N)]
// Block tile 64(M) x 32(N); 8 waves, each a 16x16 WMMA tile. K step 32.
// Requires M%64==0, N%32==0, K%32==0 (true for both call sites).
__global__ __launch_bounds__(256)
void wmma_gemm_f16(const float* __restrict__ A, const float* __restrict__ Bm,
                   const float* __restrict__ bias, float* __restrict__ C,
                   int M, int N, int K) {
    // Fragment-order LDS: [tile][lane][e] halves; per-lane 32B, 16B-aligned.
    __shared__ _Float16 Asf[4][32][16];   // 4 KB: four 16x32 A tiles
    __shared__ _Float16 Bsf[2][32][16];   // 2 KB: two 32x16 B tiles
    const int tid  = threadIdx.x;
    const int wave = tid >> 5;
    const int lane = tid & 31;
    const int mw = wave >> 1;             // 0..3 (M sub-tile)
    const int nw = wave & 1;              // 0..1 (N sub-tile)
    const int m0 = blockIdx.x * 64;
    const int n0 = blockIdx.y * 32;

    v8f acc = {};
    for (int k0 = 0; k0 < K; k0 += 32) {
        // --- stage A tile (64 rows x 32 k) as packed half2 in fragment order
        // pair index: r in [0,64), cp in [0,16) -> c = 2*cp ; 1024 pairs / 256 thr
#pragma unroll
        for (int ii = 0; ii < 4; ++ii) {
            const int i  = tid + ii * 256;
            const int r  = i >> 4;
            const int c  = (i & 15) * 2;
            const float2 av = *(const float2*)&A[(size_t)(m0 + r) * K + k0 + c];
            const int m    = r & 15;
            const int tile = r >> 4;
            const int hi   = (c >> 3) & 1;                 // k bit3
            const int ln   = m + 16 * hi;
            const int e    = c - 8 * hi - 8 * (c >> 4);    // even
            *(h2*)&Asf[tile][ln][e] = (h2){(_Float16)av.x, (_Float16)av.y};
        }
        // --- stage B tile (32 k x 32 n): kr pair x col ; 512 pairs / 256 thr
#pragma unroll
        for (int ii = 0; ii < 2; ++ii) {
            const int i  = tid + ii * 256;
            const int kr = (i >> 5) * 2;
            const int c  = i & 31;
            const float b0 = Bm[(size_t)(k0 + kr) * N + n0 + c];
            const float b1 = Bm[(size_t)(k0 + kr + 1) * N + n0 + c];
            const int ln   = (c & 15) + 16 * ((kr >> 3) & 1);
            const int tile = c >> 4;
            const int e    = kr - 8 * ((kr >> 3) & 1) - 8 * (kr >> 4); // even
            *(h2*)&Bsf[tile][ln][e] = (h2){(_Float16)b0, (_Float16)b1};
        }
        __syncthreads();

        const v16h af = *(const v16h*)&Asf[mw][lane][0];   // 2x ds_load_b128
        const v16h bf = *(const v16h*)&Bsf[nw][lane][0];
        acc = __builtin_amdgcn_wmma_f32_16x16x32_f16(
            false, af, false, bf, (short)0, acc, false, false);
        __syncthreads();
    }

    // D layout: VGPR j -> row j + 8*(lane>=16), col lane&15
    const int n  = n0 + nw * 16 + (lane & 15);
    const int mb = m0 + mw * 16 + ((lane < 16) ? 0 : 8);
    const float bv = bias ? bias[n] : 0.0f;
#pragma unroll
    for (int j = 0; j < 8; ++j)
        C[(size_t)(mb + j) * N + n] = acc[j] + bv;
}

// ---------------- K3: clustering + cosine mixing. One block per batch-head.
// Deterministic (no atomics): fixed-order LDS reductions.
__global__ __launch_bounds__(64)
void cluster_kernel(const float* __restrict__ y8,
                    const float* __restrict__ alpha,
                    const float* __restrict__ beta,
                    float* __restrict__ out8) {
    __shared__ float pa[HDIM][64];
    __shared__ float ka[2][HDIM][64];
    __shared__ float va[2][HDIM][64];
    __shared__ float kc[2][4][HDIM];
    __shared__ float vc[2][4][HDIM];
    __shared__ float agg[8][HDIM];
    __shared__ float wsoft[4][64];
    __shared__ int   hardm[64];
    __shared__ float wrow[8];
    __shared__ float anorm[8];

    const int bh = blockIdx.x;        // 0..63 = b*HEADS + head
    const int b = bh >> 2, head = bh & 3;
    const int n = threadIdx.x;        // agent 0..63
    const float* __restrict__ row = y8 + ((size_t)b * 64 + n) * NSD + head * HDIM;
#pragma unroll
    for (int c = 0; c < HDIM; ++c) {
        pa[c][n]    = row[c];            // stream 0 (p)
        ka[0][c][n] = row[96 + c];       // module0 k
        va[0][c][n] = row[192 + c];      // module0 v
        ka[1][c][n] = row[288 + c];      // module1 k
        va[1][c][n] = row[384 + c];      // module1 v
    }
    __syncthreads();

    // cluster centers: 2x2 pool of 8x8 agents (= exact 2x2 pool of 64x64)
    for (int idx = n; idx < 2 * 4 * HDIM; idx += 64) {
        const int i = idx / (4 * HDIM);
        const int m = (idx / HDIM) & 3;
        const int c = idx % HDIM;
        const int my = m >> 1, mx = m & 1;
        float sk = 0.f, sv = 0.f;
        for (int ry = 0; ry < 4; ++ry)
            for (int rx = 0; rx < 4; ++rx) {
                const int a = (my * 4 + ry) * 8 + mx * 4 + rx;
                sk += ka[i][c][a];
                sv += va[i][c][a];
            }
        kc[i][m][c] = sk * (1.f / 16.f);
        vc[i][m][c] = sv * (1.f / 16.f);
    }
    __syncthreads();

    // module 0: hard argmax assignment (first max wins, like jnp.argmax)
    {
        float best = -1e30f; int bm = 0;
        for (int m = 0; m < 4; ++m) {
            float d2 = 0.f;
            for (int c = 0; c < HDIM; ++c) {
                float d = kc[0][m][c] - ka[0][c][n]; d2 += d * d;
            }
            const float sim = -d2;
            if (sim > best) { best = sim; bm = m; }
        }
        hardm[n] = bm;
    }
    // module 1: soft (softmax over 4 clusters)
    {
        float sims[4], mx = -1e30f;
        for (int m = 0; m < 4; ++m) {
            float d2 = 0.f;
            for (int c = 0; c < HDIM; ++c) {
                float d = kc[1][m][c] - ka[1][c][n]; d2 += d * d;
            }
            sims[m] = -d2;
            mx = fmaxf(mx, sims[m]);
        }
        float se = 0.f;
        for (int m = 0; m < 4; ++m) { sims[m] = __expf(sims[m] - mx); se += sims[m]; }
        for (int m = 0; m < 4; ++m) wsoft[m][n] = sims[m] / se;
    }
    __syncthreads();

    // aggregation agg[m][c] = vc + sum_n w[m][n]*va[n][c]
    for (int idx = n; idx < 2 * 4 * HDIM; idx += 64) {
        const int i = idx / (4 * HDIM);
        const int m = (idx / HDIM) & 3;
        const int c = idx % HDIM;
        float s = 0.f;
        if (i == 0) { for (int a = 0; a < 64; ++a) s += (hardm[a] == m) ? va[0][c][a] : 0.f; }
        else        { for (int a = 0; a < 64; ++a) s += wsoft[m][a] * va[1][c][a]; }
        agg[i * 4 + m][c] = vc[i][m][c] + s;
    }
    if (n < 8) {
        const int i = n >> 2, m = n & 3;
        float ws = 0.f;
        if (i == 0) { for (int a = 0; a < 64; ++a) ws += (hardm[a] == m) ? 1.f : 0.f; }
        else        { for (int a = 0; a < 64; ++a) ws += wsoft[m][a]; }
        wrow[n] = 1.f + ws;
    }
    __syncthreads();
    for (int idx = n; idx < 8 * HDIM; idx += 64)
        agg[idx / HDIM][idx % HDIM] /= wrow[idx / HDIM];
    __syncthreads();
    if (n < 8) {
        float s = 0.f;
        for (int c = 0; c < HDIM; ++c) s += agg[n][c] * agg[n][c];
        anorm[n] = sqrtf(s) + 1e-6f;
    }
    __syncthreads();

    // cosine sim vs pa, softmax over 8 clusters, mix
    float pn = 0.f;
    for (int c = 0; c < HDIM; ++c) pn += pa[c][n] * pa[c][n];
    pn = sqrtf(pn) + 1e-6f;
    float sims[8], mx2 = -1e30f;
    for (int m = 0; m < 8; ++m) {
        float dot = 0.f;
        for (int c = 0; c < HDIM; ++c) dot += agg[m][c] * pa[c][n];
        sims[m] = alpha[m] * (dot / (anorm[m] * pn)) + beta[m];
        mx2 = fmaxf(mx2, sims[m]);
    }
    float se = 0.f;
    for (int m = 0; m < 8; ++m) { sims[m] = __expf(sims[m] - mx2); se += sims[m]; }
    const float inv = 1.f / se;
    float* __restrict__ dst = out8 + ((size_t)b * 64 + n) * DDIM + head * HDIM;
    for (int c = 0; c < HDIM; ++c) {
        float o = 0.f;
        for (int m = 0; m < 8; ++m) o += agg[m][c] * sims[m];
        dst[c] = o * inv;
    }
}

// ---------------- K5: bilinear 8x8 -> 64x64 upsample + bias, 4 outputs/thread
__global__ __launch_bounds__(256)
void upsample_kernel(const float* __restrict__ z8, const float* __restrict__ b2,
                     float* __restrict__ out) {
    const int t = blockIdx.x * 256 + threadIdx.x;   // over B*384*64*16 quads
    const int wq = t & 15;                          // 4-wide column quad
    const int h  = (t >> 4) & 63;
    const int bo = t >> 10;                         // b*384 + o
    const int o = bo % CIN;
    const int b = bo / CIN;
    const float sy = (h + 0.5f) * 0.125f - 0.5f;
    const int y0 = (int)floorf(sy);
    const float fy = sy - (float)y0;
    const int y0c = min(7, max(0, y0)), y1c = min(7, max(0, y0 + 1));
    const float* __restrict__ zb = z8 + (size_t)b * 64 * CIN + o;
    const float bias = b2[o];
    float rv[4];
#pragma unroll
    for (int j = 0; j < 4; ++j) {
        const int w = wq * 4 + j;
        const float sx = (w + 0.5f) * 0.125f - 0.5f;
        const int x0 = (int)floorf(sx);
        const float fx = sx - (float)x0;
        const int x0c = min(7, max(0, x0)), x1c = min(7, max(0, x0 + 1));
        const float z00 = zb[(y0c * 8 + x0c) * CIN];
        const float z01 = zb[(y0c * 8 + x1c) * CIN];
        const float z10 = zb[(y1c * 8 + x0c) * CIN];
        const float z11 = zb[(y1c * 8 + x1c) * CIN];
        rv[j] = (1.f - fy) * ((1.f - fx) * z00 + fx * z01)
              +        fy  * ((1.f - fx) * z10 + fx * z11) + bias;
    }
    ((float4*)out)[t] = make_float4(rv[0], rv[1], rv[2], rv[3]);
}

// ---------------------------------------------------------------------------
extern "C" void kernel_launch(void* const* d_in, const int* in_sizes, int n_in,
                              void* d_out, int out_size, void* d_ws, size_t ws_size,
                              hipStream_t stream) {
    (void)in_sizes; (void)n_in; (void)out_size; (void)ws_size;
    const float* x  = (const float*)d_in[0];
    const float* W1 = (const float*)d_in[1];
    const float* b1 = (const float*)d_in[2];
    const float* W2 = (const float*)d_in[3];
    const float* b2 = (const float*)d_in[4];
    const float* sa = (const float*)d_in[5];
    const float* sb = (const float*)d_in[6];
    float* out = (float*)d_out;

    float* ws = (float*)d_ws;
    float* x8 = ws;                       // 1024 x 384
    float* y8 = x8 + 1024 * CIN;          // 1024 x 480
    float* o8 = y8 + 1024 * NSD;          // 1024 x 96
    float* z8 = o8 + 1024 * DDIM;         // 1024 x 384

    pool_x_kernel<<<16 * CIN, 256, 0, stream>>>(x, x8);

    dim3 g1(1024 / 64, NSD / 32);
    wmma_gemm_f16<<<g1, 256, 0, stream>>>(x8, W1, b1, y8, 1024, NSD, CIN);

    cluster_kernel<<<64, 64, 0, stream>>>(y8, sa, sb, o8);

    dim3 g2(1024 / 64, CIN / 32);
    wmma_gemm_f16<<<g2, 256, 0, stream>>>(o8, W2, nullptr, z8, 1024, CIN, DDIM);

    upsample_kernel<<<(16 * CIN * 64 * 16) / 256, 256, 0, stream>>>(z8, b2, out);
}
